// GATNet_867583394114
// MI455X (gfx1250) — matrix operand order repaired and checked by reading, not compiled, not run on hardware
//
#include <hip/hip_runtime.h>
#include <hip/hip_bf16.h>

// ---------------------------------------------------------------------------
// GATNet for gfx1250 (MI455X).
//  * GEMMs: v_wmma_f32_16x16x32_bf16, wave32, each wave -> 16x64 output strip
//    (4 accumulators, 4x A-fragment reuse). A panel staged into LDS with
//    global_load_async_to_lds_b128 (ASYNCcnt) from a bf16 activation mirror.
//    B fragments are double-buffered with a manual 2x-unrolled k-loop
//    (register role swap, no copies): loads for kt+1 overlap WMMAs of kt.
//    Wb carries one extra k-step of padding so the prefetch is branchless.
//  * Edge softmax + aggregation: float atomics (L2-resident feature matrices).
//
// WMMA register-layout assumptions (CDNA5 ISA §7.12.2):
//   A (16x32 bf16): lane l -> row M = l&15, khalf = l>>4;
//                   elements 0..7  = K khalf*8 + 0..7
//                   elements 8..15 = K khalf*8 + 16..23
//   B (32x16 bf16): lane l -> col N = l&15, khalf = l>>4;
//                   same K order as A               (pre-packed by pack_w)
//   C/D (16x16 f32): VGPR r, lanes 0-15 -> (M=r, N=lane);
//                    lanes 16-31 -> (M=r+8, N=lane-16)
// ---------------------------------------------------------------------------

typedef __attribute__((ext_vector_type(16))) __bf16       v16bf;
typedef __attribute__((ext_vector_type(8)))  float        v8f;
typedef __attribute__((ext_vector_type(4)))  unsigned int v4u;

union FragU { v16bf bf; v4u q[2]; };

#define NEG_SLOPE 0.2f

__device__ inline unsigned short f2bf(float f) {
  unsigned int u = __float_as_uint(f);
  u += 0x7FFFu + ((u >> 16) & 1u);           // round-to-nearest-even
  return (unsigned short)(u >> 16);
}

__device__ inline float leaky(float v) { return v > 0.f ? v : NEG_SLOPE * v; }

// Monotonic-mapping float atomic max via integer atomics.
__device__ inline void atomicMaxF32(float* addr, float val) {
  if (val >= 0.0f) atomicMax((int*)addr, __float_as_int(val));
  else             atomicMin((unsigned int*)addr, (unsigned int)__float_as_int(val));
}

// Load 4 consecutive n-tile B fragments (8x b128) for one k-step.
__device__ inline void load_b4(FragU b[4], const unsigned short* p) {
#pragma unroll
  for (int j = 0; j < 4; ++j) {
    const v4u* bp = (const v4u*)(p + (size_t)j * 512);
    b[j].q[0] = bp[0];
    b[j].q[1] = bp[1];
  }
}

// ---------------------------------------------------------------------------
// fp32 -> bf16 mirror of activations (GEMM A operand lives as bf16 in global)
// ---------------------------------------------------------------------------
__global__ void f32_to_bf16(const float* __restrict__ src,
                            unsigned short* __restrict__ dst, int n) {
  int i = blockIdx.x * blockDim.x + threadIdx.x;
  if (i < n) dst[i] = f2bf(src[i]);
}

// ReLU in place + bf16 mirror for the next layer's GEMM
__global__ void relu_bf16(float* __restrict__ x,
                          unsigned short* __restrict__ xbf, int n) {
  int i = blockIdx.x * blockDim.x + threadIdx.x;
  if (i < n) {
    float v = fmaxf(x[i], 0.f);
    x[i]   = v;
    xbf[i] = f2bf(v);
  }
}

// ---------------------------------------------------------------------------
// Pack W [din x dout] fp32 row-major -> bf16 WMMA-B fragments.
// Wb[((kt*numNt + nt)*32 + lane)*16 + e] =
//     bf16(W[(kt*32 + (e&8)*2 + (lane>>4)*8 + (e&7)), nt*16 + (lane&15)])
// ---------------------------------------------------------------------------
__global__ void pack_w_kernel(const float* __restrict__ W,
                              unsigned short* __restrict__ Wb,
                              int din, int dout) {
  int t = blockIdx.x * blockDim.x + threadIdx.x;
  if (t >= din * dout) return;
  int e     = t & 15;
  int lane  = (t >> 4) & 31;
  int frag  = t >> 9;
  int numNt = dout >> 4;
  int nt    = frag % numNt;
  int kt    = frag / numNt;
  int k = kt * 32 + (e & 8) * 2 + (lane >> 4) * 8 + (e & 7);
  int n = nt * 16 + (lane & 15);
  Wb[t] = f2bf(W[(size_t)k * dout + n]);
}

// ---------------------------------------------------------------------------
// H = A @ W, A is bf16 [N x din] row-major, W pre-packed bf16 fragments.
// Block: 256 thr = 8 waves. Each wave -> one 16x64 strip (4 WMMA tiles).
//   wptile = dout/64 waves per 16-row tile; rowsPB = (8/wptile)*16 rows/block.
// A panel staged to LDS via global_load_async_to_lds_b128 (ASYNCcnt).
// Requires numKt (= din/32) to be even, and Wb padded by one k-step.
// ---------------------------------------------------------------------------
__global__ __launch_bounds__(256)
void gemm_bf16_wmma(const unsigned short* __restrict__ Abf,
                    const unsigned short* __restrict__ Wb,
                    float* __restrict__ H, int din, int dout) {
  __shared__ unsigned short sA[64 * 256];      // up to 32 KB (64 rows x 256 K)
  const int tid    = threadIdx.x;
  const int lane   = tid & 31;
  const int wave   = tid >> 5;
  const int wptile = dout >> 6;                // 2 (dout=128) or 4 (dout=256)
  const int rtiles = 8 / wptile;               // 16-row tiles per block
  const int rowsPB = rtiles << 4;
  const int row0   = blockIdx.x * rowsPB;
  const int rt     = wave / wptile;            // this wave's row tile
  const int nblk   = (wave % wptile) << 6;     // this wave's starting column
  const int numNt  = dout >> 4;

  // ---- stage A rows (contiguous bf16 block) into LDS ----
  const int chunks = (rowsPB * din) >> 3;      // 16-byte chunks
  const unsigned short* gsrc = Abf + (size_t)row0 * din;
#if defined(__HIP_DEVICE_COMPILE__) && defined(__gfx1250__)
  for (int i = tid; i < chunks; i += 256) {
    unsigned int loff = (unsigned int)(uintptr_t)(&sA[i << 3]);  // LDS offset
    unsigned long long gaddr = (unsigned long long)(uintptr_t)(gsrc + ((size_t)i << 3));
    asm volatile("global_load_async_to_lds_b128 %0, %1, off"
                 :: "v"(loff), "v"(gaddr) : "memory");
  }
  asm volatile("s_wait_asynccnt 0x0" ::: "memory");
#else
  for (int i = tid; i < chunks; i += 256)
    ((v4u*)sA)[i] = ((const v4u*)gsrc)[i];
#endif
  __syncthreads();

  const int khalf = lane >> 4;                 // 0 or 1
  const int msel  = lane & 15;                 // A row / B,C column
  const int numKt = din >> 5;                  // even (4 or 8)
  const unsigned short* sArow = &sA[(rt * 16 + msel) * din];

  // Per-lane base of this wave's first B fragment; one kt step advances by
  // numNt fragments = numNt*512 ushorts. n-tile j is +j*512 ushorts.
  const unsigned short* bbase = &Wb[(((size_t)(nblk >> 4)) * 32 + lane) * 16];
  const size_t ktStride = (size_t)numNt * 512;

  v8f acc[4] = {{}, {}, {}, {}};
  FragU b0[4], b1[4];
  load_b4(b0, bbase);                          // kt = 0

  // Manual 2x-unrolled double buffer: no register copies, no branches.
  // The kt+2 prefetch on the final iteration reads Wb's padding (unused).
  for (int kt = 0; kt < numKt; kt += 2) {
    load_b4(b1, bbase + (size_t)(kt + 1) * ktStride);
    {
      const int k0 = kt * 32 + khalf * 8;
      FragU a;
      a.q[0] = *(const v4u*)&sArow[k0];        // K = k0 .. k0+7
      a.q[1] = *(const v4u*)&sArow[k0 + 16];   // K = k0+16 .. k0+23
#pragma unroll
      for (int j = 0; j < 4; ++j)
        acc[j] = __builtin_amdgcn_wmma_f32_16x16x32_bf16(
            false, a.bf, false, b0[j].bf, (short)0, acc[j], false, false);
    }
    load_b4(b0, bbase + (size_t)(kt + 2) * ktStride);
    {
      const int k0 = (kt + 1) * 32 + khalf * 8;
      FragU a;
      a.q[0] = *(const v4u*)&sArow[k0];
      a.q[1] = *(const v4u*)&sArow[k0 + 16];
#pragma unroll
      for (int j = 0; j < 4; ++j)
        acc[j] = __builtin_amdgcn_wmma_f32_16x16x32_bf16(
            false, a.bf, false, b1[j].bf, (short)0, acc[j], false, false);
    }
  }

  const int rbase = row0 + rt * 16 + khalf * 8;
#pragma unroll
  for (int j = 0; j < 4; ++j) {
    const int col = nblk + j * 16 + msel;
#pragma unroll
    for (int r = 0; r < 8; ++r)
      H[(size_t)(rbase + r) * dout + col] = acc[j][r];
  }
}

// ---------------------------------------------------------------------------
// al_s[n] = H[n,:].a_src ; al_d[n] = H[n,:].a_dst  (one wave per node)
// ---------------------------------------------------------------------------
__global__ void node_dots(const float* __restrict__ H,
                          const float* __restrict__ avs,
                          const float* __restrict__ avd,
                          float* __restrict__ als, float* __restrict__ ald,
                          int dout) {
  int node = blockIdx.x * 8 + (threadIdx.x >> 5);
  int lane = threadIdx.x & 31;
  float s = 0.f, d = 0.f;
  for (int j = lane; j < dout; j += 32) {
    float h = H[(size_t)node * dout + j];
    s = fmaf(h, avs[j], s);
    d = fmaf(h, avd[j], d);
  }
#pragma unroll
  for (int m = 16; m; m >>= 1) {
    s += __shfl_xor(s, m, 32);
    d += __shfl_xor(d, m, 32);
  }
  if (lane == 0) { als[node] = s; ald[node] = d; }
}

__global__ void init_softmax(float* __restrict__ m, float* __restrict__ denom, int N) {
  int i = blockIdx.x * blockDim.x + threadIdx.x;
  if (i < N) { m[i] = __uint_as_float(0xFF800000u); denom[i] = 0.f; }
}

__global__ void edge_max(const float* __restrict__ als, const float* __restrict__ ald,
                         const int* __restrict__ esrc, const int* __restrict__ edst,
                         int E, int Etot, float* __restrict__ m) {
  int i = blockIdx.x * blockDim.x + threadIdx.x;
  if (i >= Etot) return;
  int s = (i < E) ? esrc[i] : (i - E);     // implicit self-loops after E
  int d = (i < E) ? edst[i] : (i - E);
  atomicMaxF32(&m[d], leaky(als[s] + ald[d]));
}

__global__ void edge_exp(const float* __restrict__ als, const float* __restrict__ ald,
                         const int* __restrict__ esrc, const int* __restrict__ edst,
                         int E, int Etot, const float* __restrict__ m,
                         float* __restrict__ exv, float* __restrict__ denom) {
  int i = blockIdx.x * blockDim.x + threadIdx.x;
  if (i >= Etot) return;
  int s = (i < E) ? esrc[i] : (i - E);
  int d = (i < E) ? edst[i] : (i - E);
  float e  = leaky(als[s] + ald[d]);
  float ex = __expf(e - m[d]);
  exv[i] = ex;
  atomicAdd(&denom[d], ex);
}

__global__ void init_bias(float* __restrict__ out, const float* __restrict__ b,
                          int n, int dout) {
  int i = blockIdx.x * blockDim.x + threadIdx.x;
  if (i < n) out[i] = b[i & (dout - 1)];   // dout is a power of two (128/256)
}

// out[dst] += alpha * h[src]; one thread per (edge, 4-float chunk)
__global__ void aggregate(const float* __restrict__ Hpre,
                          const float* __restrict__ exv,
                          const float* __restrict__ denom,
                          const int* __restrict__ esrc, const int* __restrict__ edst,
                          int E, int Etot, float* __restrict__ out, int dout) {
  int t = blockIdx.x * blockDim.x + threadIdx.x;
  int chunks = dout >> 2;
  int total  = Etot * chunks;
  if (t >= total) return;
  int edge = t / chunks;
  int c    = t - edge * chunks;
  int s = (edge < E) ? esrc[edge] : (edge - E);
  int d = (edge < E) ? edst[edge] : (edge - E);
  float alpha = exv[edge] / denom[d];
  const float4 h = *(const float4*)&Hpre[(size_t)s * dout + c * 4];
  float* o = &out[(size_t)d * dout + c * 4];
  atomicAdd(o + 0, alpha * h.x);
  atomicAdd(o + 1, alpha * h.y);
  atomicAdd(o + 2, alpha * h.z);
  atomicAdd(o + 3, alpha * h.w);
}

// ---------------------------------------------------------------------------
// Host orchestration
// ---------------------------------------------------------------------------
static void run_layer(const unsigned short* Abf, int din, int dout,
                      const float* W, const float* avs, const float* avd,
                      const float* bias, float* Hpre, float* out,
                      unsigned short* Wb, float* als, float* ald,
                      float* mbuf, float* denom, float* exbuf,
                      const int* esrc, const int* edst,
                      int E, int Etot, int N,
                      unsigned short* next_Abf /* NULL => no relu */,
                      hipStream_t stream) {
  const int BT = 256;
  // 1) pack W -> bf16 B-fragments
  pack_w_kernel<<<(din * dout + BT - 1) / BT, BT, 0, stream>>>(W, Wb, din, dout);
  // 2) GEMM: Hpre = Abf @ W
  const int wptile = dout >> 6;
  const int rowsPB = (8 / wptile) << 4;
  gemm_bf16_wmma<<<N / rowsPB, BT, 0, stream>>>(Abf, Wb, Hpre, din, dout);
  // 3) per-node attention dots
  node_dots<<<N / 8, BT, 0, stream>>>(Hpre, avs, avd, als, ald, dout);
  // 4) scatter softmax
  init_softmax<<<(N + BT - 1) / BT, BT, 0, stream>>>(mbuf, denom, N);
  edge_max<<<(Etot + BT - 1) / BT, BT, 0, stream>>>(als, ald, esrc, edst, E, Etot, mbuf);
  edge_exp<<<(Etot + BT - 1) / BT, BT, 0, stream>>>(als, ald, esrc, edst, E, Etot,
                                                    mbuf, exbuf, denom);
  // 5) aggregate (+ bias init)
  init_bias<<<(N * dout + BT - 1) / BT, BT, 0, stream>>>(out, bias, N * dout, dout);
  int total = Etot * (dout >> 2);
  aggregate<<<(total + BT - 1) / BT, BT, 0, stream>>>(Hpre, exbuf, denom, esrc, edst,
                                                      E, Etot, out, dout);
  // 6) ReLU + bf16 mirror for next layer
  if (next_Abf)
    relu_bf16<<<(N * dout + BT - 1) / BT, BT, 0, stream>>>(out, next_Abf, N * dout);
}

extern "C" void kernel_launch(void* const* d_in, const int* in_sizes, int n_in,
                              void* d_out, int out_size, void* d_ws, size_t ws_size,
                              hipStream_t stream) {
  const int DIN = 128, DH = 256, DOUT = 128;
  const float* x   = (const float*)d_in[0];
  const int*   ei  = (const int*)d_in[1];
  const float* W1  = (const float*)d_in[3];
  const float* as1 = (const float*)d_in[4];
  const float* ad1 = (const float*)d_in[5];
  const float* b1  = (const float*)d_in[6];
  const float* W2  = (const float*)d_in[7];
  const float* as2 = (const float*)d_in[8];
  const float* ad2 = (const float*)d_in[9];
  const float* b2  = (const float*)d_in[10];
  const float* W3  = (const float*)d_in[11];
  const float* as3 = (const float*)d_in[12];
  const float* ad3 = (const float*)d_in[13];
  const float* b3  = (const float*)d_in[14];

  const int N    = in_sizes[0] / DIN;   // 40000
  const int E    = in_sizes[1] / 2;     // 400000
  const int Etot = E + N;               // + self loops
  const int* esrc = ei;
  const int* edst = ei + E;
  const int BT = 256;

  // Carve workspace
  char* w = (char*)d_ws;
  float* Hpre = (float*)w;                  w += (size_t)N * DH * sizeof(float);
  float* Hagg = (float*)w;                  w += (size_t)N * DH * sizeof(float);
  unsigned short* Abf = (unsigned short*)w; w += (size_t)N * DH * sizeof(unsigned short);
  // Wb padded by one k-step (numNt*512 <= 8192 ushorts) for branchless prefetch
  unsigned short* Wb  = (unsigned short*)w; w += ((size_t)DH * DH + 8192) * sizeof(unsigned short);
  float* als   = (float*)w;                 w += (size_t)N * sizeof(float);
  float* ald   = (float*)w;                 w += (size_t)N * sizeof(float);
  float* mbuf  = (float*)w;                 w += (size_t)N * sizeof(float);
  float* denom = (float*)w;                 w += (size_t)N * sizeof(float);
  float* exbuf = (float*)w;                 /* Etot floats */

  // bf16 mirror of x for layer-1 GEMM
  f32_to_bf16<<<(N * DIN + BT - 1) / BT, BT, 0, stream>>>(x, Abf, N * DIN);

  // Layer 1: [N,128] -> Hagg[N,256], ReLU (+ bf16 mirror into Abf)
  run_layer(Abf, DIN, DH, W1, as1, ad1, b1, Hpre, Hagg, Wb, als, ald,
            mbuf, denom, exbuf, esrc, edst, E, Etot, N, Abf, stream);
  // Layer 2: [N,256] -> Hagg[N,256], ReLU (+ bf16 mirror into Abf)
  run_layer(Abf, DH, DH, W2, as2, ad2, b2, Hpre, Hagg, Wb, als, ald,
            mbuf, denom, exbuf, esrc, edst, E, Etot, N, Abf, stream);
  // Layer 3: [N,256] -> d_out[N,128], no ReLU
  run_layer(Abf, DH, DOUT, W3, as3, ad3, b3, Hpre, (float*)d_out, Wb, als, ald,
            mbuf, denom, exbuf, esrc, edst, E, Etot, N, nullptr, stream);
}
